// MixtureOfExperts_47278999994874
// MI455X (gfx1250) — compile-verified, matrix operand order checked by reading, not run on hardware
//
#include <hip/hip_runtime.h>
#include <hip/hip_bf16.h>

#define T_TOK 8192
#define DDIM  1024
#define EEXP  8
#define HDIM  3072
#define NSLOT (2 * T_TOK)
#define KS1   (DDIM / 64)
#define KS2   (HDIM / 64)

typedef __attribute__((ext_vector_type(16))) __bf16    v16bf;
typedef __attribute__((ext_vector_type(8)))  float     v8f;
typedef __attribute__((ext_vector_type(4)))  unsigned  v4u;
typedef __attribute__((ext_vector_type(8)))  unsigned  v8u;

union Frag { v16bf v; uint4 q[2]; };

static __device__ __forceinline__ unsigned short f2bf(float f) {
    union { float f; unsigned u; } c; c.f = f;
    unsigned r = c.u + 0x7FFFu + ((c.u >> 16) & 1u);   // round-to-nearest-even
    return (unsigned short)(r >> 16);
}

static __device__ __forceinline__ v8f vzero8() {
    v8f z = {0.f, 0.f, 0.f, 0.f, 0.f, 0.f, 0.f, 0.f};
    return z;
}

static __device__ __forceinline__ unsigned ldsoff(const void* p) {
    // flat LDS aperture: addr[31:0] is the wave-relative LDS byte offset
    return (unsigned)(unsigned long long)(uintptr_t)p;
}

// Issue one TDM copy of a (rows x 64) bf16 tile (tensor row pitch = tdim0
// elements) from global memory into LDS at byte offset lds_off, padding each
// 128B tile row to a 144B LDS pitch (pad_interval=4 -> 128B, pad_amount=3 -> 16B).
static __device__ __forceinline__ void tdm_load_tile(
    unsigned lds_off, const unsigned short* gsrc,
    unsigned tdim0, unsigned tdim1, unsigned rows)
{
    unsigned long long ga = (unsigned long long)(uintptr_t)gsrc;
    v4u g0;
    g0[0] = 1u;                                          // count=1, user-mode, no gather
    g0[1] = lds_off;                                     // LDS byte address
    g0[2] = (unsigned)ga;                                // global_addr[31:0]
    g0[3] = (unsigned)(ga >> 32) | (2u << 30);           // global_addr[56:32] | type=2
    v8u g1;
    g1[0] = (1u << 16) | (1u << 20) | (4u << 22) | (3u << 25); // data=2B, pad_en, int, amt
    g1[1] = (tdim0 & 0xFFFFu) << 16;                     // tensor_dim0[15:0]
    g1[2] = (tdim0 >> 16) | ((tdim1 & 0xFFFFu) << 16);   // dim0[31:16] | dim1[15:0]
    g1[3] = (tdim1 >> 16) | (64u << 16);                 // dim1[31:16] | tile_dim0=64
    g1[4] = rows;                                        // tile_dim1=rows, tile_dim2=0
    g1[5] = tdim0;                                       // tensor_dim0_stride[31:0]
    g1[6] = 0u;
    g1[7] = 0u;
    asm volatile("tensor_load_to_lds %0, %1" :: "s"(g0), "s"(g1) : "memory");
}

// ---------------------------------------------------------------- kernel 0
__global__ void moe_zero_cnt(int* cnt) {
    if (threadIdx.x < EEXP) cnt[threadIdx.x] = 0;
}

// ---------------------------------------------------------------- gating + top-2 routing (one wave32 per token)
__global__ __launch_bounds__(256) void moe_gate(
    const float* __restrict__ x, const float* __restrict__ gw,
    int* __restrict__ cnt, int* __restrict__ tok_list,
    int* __restrict__ slot_list, float* __restrict__ gw_list)
{
    const int lane = threadIdx.x & 31;
    const int wv   = threadIdx.x >> 5;
    const int t    = blockIdx.x * 8 + wv;

    float acc[EEXP];
#pragma unroll
    for (int e = 0; e < EEXP; ++e) acc[e] = 0.f;

    const float* xr = x + (size_t)t * DDIM;
    for (int j = 0; j < DDIM / 32; ++j) {
        int d = j * 32 + lane;
        float xv = xr[d];
        const float4* g = (const float4*)(gw + (size_t)d * EEXP);
        float4 g0 = g[0], g1 = g[1];
        acc[0] += xv * g0.x; acc[1] += xv * g0.y;
        acc[2] += xv * g0.z; acc[3] += xv * g0.w;
        acc[4] += xv * g1.x; acc[5] += xv * g1.y;
        acc[6] += xv * g1.z; acc[7] += xv * g1.w;
    }
#pragma unroll
    for (int e = 0; e < EEXP; ++e)
#pragma unroll
        for (int off = 16; off > 0; off >>= 1)
            acc[e] += __shfl_xor(acc[e], off, 32);

    if (lane == 0) {
        int i0 = 0;
#pragma unroll
        for (int e = 1; e < EEXP; ++e) if (acc[e] > acc[i0]) i0 = e;
        int i1 = (i0 == 0) ? 1 : 0;
#pragma unroll
        for (int e = 0; e < EEXP; ++e) if (e != i0 && acc[e] > acc[i1]) i1 = e;
        float p0 = 1.f;                              // exp(l0 - l0)
        float p1 = __expf(acc[i1] - acc[i0]);
        float rs = __builtin_amdgcn_rcpf(p0 + p1);
        float w0 = p0 * rs, w1v = p1 * rs;

        int p = atomicAdd(&cnt[i0], 1);
        tok_list [i0 * T_TOK + p] = t;
        slot_list[i0 * T_TOK + p] = 2 * t;
        gw_list  [i0 * T_TOK + p] = w0;

        p = atomicAdd(&cnt[i1], 1);
        tok_list [i1 * T_TOK + p] = t;
        slot_list[i1 * T_TOK + p] = 2 * t + 1;
        gw_list  [i1 * T_TOK + p] = w1v;
    }
}

// ---------------------------------------------------------------- x fp32 -> bf16
__global__ __launch_bounds__(256) void moe_x2bf(
    const float* __restrict__ x, unsigned short* __restrict__ xb)
{
    size_t base = ((size_t)blockIdx.x * 256 + threadIdx.x) * 8;
    const float4* p = (const float4*)(x + base);
    float4 a = p[0], b = p[1];
    uint4 o;
    o.x = (unsigned)f2bf(a.x) | ((unsigned)f2bf(a.y) << 16);
    o.y = (unsigned)f2bf(a.z) | ((unsigned)f2bf(a.w) << 16);
    o.z = (unsigned)f2bf(b.x) | ((unsigned)f2bf(b.y) << 16);
    o.w = (unsigned)f2bf(b.z) | ((unsigned)f2bf(b.w) << 16);
    *(uint4*)(xb + base) = o;
}

// ---------------------------------------------------------------- weight convert + transpose: [E][R][C] f32 -> [E][C][R] bf16
__global__ __launch_bounds__(256) void moe_wtrans(
    const float* __restrict__ in, unsigned short* __restrict__ outw, int R, int C)
{
    __shared__ float tile[32][33];
    const size_t eoff = (size_t)blockIdx.z * R * C;
    const float* ine = in + eoff;
    unsigned short* oute = outw + eoff;
    const int c0 = blockIdx.x * 32, r0 = blockIdx.y * 32;
    const int tc = threadIdx.x & 31, tr = threadIdx.x >> 5;   // tr in 0..7
#pragma unroll
    for (int i = 0; i < 4; ++i)
        tile[tr + 8 * i][tc] = ine[(size_t)(r0 + tr + 8 * i) * C + c0 + tc];
    __syncthreads();
#pragma unroll
    for (int i = 0; i < 4; ++i)
        oute[(size_t)(c0 + tr + 8 * i) * R + r0 + tc] = f2bf(tile[tc][tr + 8 * i]);
}

// ---------------------------------------------------------------- GEMM1: h = silu(X@W1) * (X@W3); BM=64, BN=128, BK=64; 2x2 tiles/wave
__global__ __launch_bounds__(256) void moe_gemm1(
    const unsigned short* __restrict__ xb,
    const unsigned short* __restrict__ wt1, const unsigned short* __restrict__ wt3,
    const int* __restrict__ cnt, const int* __restrict__ tok_list,
    const int* __restrict__ slot_list, unsigned short* __restrict__ hbuf)
{
    const int e  = blockIdx.z;
    const int m0 = blockIdx.y * 64;
    const int n0 = blockIdx.x * 128;
    const int ce = cnt[e];
    if (m0 >= ce) return;                       // uniform early exit (graph-static grid)

    __shared__ __align__(16) unsigned short sA [2][64][72];
    __shared__ __align__(16) unsigned short sB1[2][128][72];
    __shared__ __align__(16) unsigned short sB3[2][128][72];
    __shared__ int sTok[64];
    __shared__ int sSlot[64];

    const int tid = threadIdx.x;
    if (tid < 64) {
        int i = m0 + tid;
        if (i < ce) { sTok[tid] = tok_list[e * T_TOK + i]; sSlot[tid] = slot_list[e * T_TOK + i]; }
        else        { sTok[tid] = 0;                       sSlot[tid] = -1; }
    }
    __syncthreads();

    const unsigned short* w1e = wt1 + (size_t)e * DDIM * HDIM + (size_t)n0 * DDIM; // [H][D] bf16
    const unsigned short* w3e = wt3 + (size_t)e * DDIM * HDIM + (size_t)n0 * DDIM;

    const int lane = tid & 31, wv = tid >> 5;
    const int wm = wv >> 2;            // wave m-group (0..1), owns m-tiles wm*2+{0,1}
    const int wn = wv & 3;             // wave n-group (0..3), owns n-tiles wn*2+{0,1}
    const int lh = lane & 15, sel = lane >> 4;
    const int arow = tid >> 2, achk = (tid & 3) * 16;

    const unsigned short* aSrc = xb + (size_t)sTok[arow] * DDIM + achk;

    v8f acc1[2][2], acc3[2][2];
#pragma unroll
    for (int mi = 0; mi < 2; ++mi)
#pragma unroll
        for (int j = 0; j < 2; ++j) { acc1[mi][j] = vzero8(); acc3[mi][j] = vzero8(); }

    { // prologue: stage 0 -> buffer 0
        uint4 a0 = *(const uint4*)(aSrc);
        uint4 a1 = *(const uint4*)(aSrc + 8);
        if (wv == 0) {
            tdm_load_tile(ldsoff(&sB1[0][0][0]), w1e, DDIM, HDIM, 128);
            tdm_load_tile(ldsoff(&sB3[0][0][0]), w3e, DDIM, HDIM, 128);
        }
        *(uint4*)&sA[0][arow][achk]     = a0;
        *(uint4*)&sA[0][arow][achk + 8] = a1;
        __builtin_amdgcn_s_wait_tensorcnt(0);   // no-op on waves that issued nothing
    }
    __syncthreads();

    for (int i = 0; i < KS1; ++i) {
        const int cur = i & 1, nxt = cur ^ 1;
        const bool more = (i + 1) < KS1;
        uint4 a0, a1;
        if (more) {
            const unsigned short* s = aSrc + (i + 1) * 64;
            a0 = *(const uint4*)s;
            a1 = *(const uint4*)(s + 8);
            if (wv == 0) {
                tdm_load_tile(ldsoff(&sB1[nxt][0][0]), w1e + (i + 1) * 64, DDIM, HDIM, 128);
                tdm_load_tile(ldsoff(&sB3[nxt][0][0]), w3e + (i + 1) * 64, DDIM, HDIM, 128);
            }
        }
#pragma unroll
        for (int ks = 0; ks < 2; ++ks) {
            const int kb = ks * 32 + sel * 8;
            Frag fa[2], fb1[2], fb3[2];
#pragma unroll
            for (int mi = 0; mi < 2; ++mi) {
                const int ar = (wm * 2 + mi) * 16 + lh;
                fa[mi].q[0] = *(const uint4*)&sA[cur][ar][kb];
                fa[mi].q[1] = *(const uint4*)&sA[cur][ar][kb + 16];
            }
#pragma unroll
            for (int j = 0; j < 2; ++j) {
                const int br = (wn * 2 + j) * 16 + lh;
                fb1[j].q[0] = *(const uint4*)&sB1[cur][br][kb];
                fb1[j].q[1] = *(const uint4*)&sB1[cur][br][kb + 16];
                fb3[j].q[0] = *(const uint4*)&sB3[cur][br][kb];
                fb3[j].q[1] = *(const uint4*)&sB3[cur][br][kb + 16];
            }
#pragma unroll
            for (int mi = 0; mi < 2; ++mi)
#pragma unroll
                for (int j = 0; j < 2; ++j) {
                    acc1[mi][j] = __builtin_amdgcn_wmma_f32_16x16x32_bf16(
                        false, fa[mi].v, false, fb1[j].v, (short)0, acc1[mi][j], false, false);
                    acc3[mi][j] = __builtin_amdgcn_wmma_f32_16x16x32_bf16(
                        false, fa[mi].v, false, fb3[j].v, (short)0, acc3[mi][j], false, false);
                }
        }
        if (more) {
            *(uint4*)&sA[nxt][arow][achk]     = a0;
            *(uint4*)&sA[nxt][arow][achk + 8] = a1;
        }
        __builtin_amdgcn_s_wait_tensorcnt(0);
        __syncthreads();
    }

    // SwiGLU (rcp-based), scatter-store bf16 rows by slot
#pragma unroll
    for (int mi = 0; mi < 2; ++mi)
#pragma unroll
        for (int j = 0; j < 2; ++j)
#pragma unroll
            for (int r = 0; r < 8; ++r) {
                int rr = (wm * 2 + mi) * 16 + sel * 8 + r;
                int slot = sSlot[rr];
                if (slot >= 0) {
                    float a1v = acc1[mi][j][r], a3v = acc3[mi][j][r];
                    float hv = a1v * __builtin_amdgcn_rcpf(1.f + __expf(-a1v)) * a3v;
                    hbuf[(size_t)slot * HDIM + n0 + (wn * 2 + j) * 16 + lh] = f2bf(hv);
                }
            }
}

// ---------------------------------------------------------------- GEMM2: y[slot] = gate_w * (h @ W2); BM=64, BN=128
__global__ __launch_bounds__(256) void moe_gemm2(
    const unsigned short* __restrict__ hbuf,
    const unsigned short* __restrict__ wt2,
    const int* __restrict__ cnt, const int* __restrict__ slot_list,
    const float* __restrict__ gw_list, float* __restrict__ y)
{
    const int e  = blockIdx.z;
    const int m0 = blockIdx.y * 64;
    const int n0 = blockIdx.x * 128;
    const int ce = cnt[e];
    if (m0 >= ce) return;

    __shared__ __align__(16) unsigned short sA[2][64][72];
    __shared__ __align__(16) unsigned short sB[2][128][72];
    __shared__ int   sSlot[64];
    __shared__ float sGw[64];

    const int tid = threadIdx.x;
    if (tid < 64) {
        int i = m0 + tid;
        if (i < ce) { sSlot[tid] = slot_list[e * T_TOK + i]; sGw[tid] = gw_list[e * T_TOK + i]; }
        else        { sSlot[tid] = -1;                       sGw[tid] = 0.f; }
    }
    __syncthreads();

    const unsigned short* w2e = wt2 + (size_t)e * HDIM * DDIM + (size_t)n0 * HDIM; // [D][H] bf16

    const int lane = tid & 31, wv = tid >> 5;
    const int wm = wv >> 2, wn = wv & 3;
    const int lh = lane & 15, sel = lane >> 4;
    const int arow = tid >> 2, achk = (tid & 3) * 16;

    int asl = sSlot[arow]; if (asl < 0) asl = 0;
    const unsigned short* aSrc = hbuf + (size_t)asl * HDIM + achk;

    v8f acc[2][2];
#pragma unroll
    for (int mi = 0; mi < 2; ++mi)
#pragma unroll
        for (int j = 0; j < 2; ++j) acc[mi][j] = vzero8();

    { // prologue
        uint4 a0 = *(const uint4*)(aSrc);
        uint4 a1 = *(const uint4*)(aSrc + 8);
        if (wv == 0) tdm_load_tile(ldsoff(&sB[0][0][0]), w2e, HDIM, DDIM, 128);
        *(uint4*)&sA[0][arow][achk]     = a0;
        *(uint4*)&sA[0][arow][achk + 8] = a1;
        __builtin_amdgcn_s_wait_tensorcnt(0);
    }
    __syncthreads();

    for (int i = 0; i < KS2; ++i) {
        const int cur = i & 1, nxt = cur ^ 1;
        const bool more = (i + 1) < KS2;
        uint4 a0, a1;
        if (more) {
            const unsigned short* s = aSrc + (i + 1) * 64;
            a0 = *(const uint4*)s;
            a1 = *(const uint4*)(s + 8);
            if (wv == 0)
                tdm_load_tile(ldsoff(&sB[nxt][0][0]), w2e + (i + 1) * 64, HDIM, DDIM, 128);
        }
#pragma unroll
        for (int ks = 0; ks < 2; ++ks) {
            const int kb = ks * 32 + sel * 8;
            Frag fa[2], fb[2];
#pragma unroll
            for (int mi = 0; mi < 2; ++mi) {
                const int ar = (wm * 2 + mi) * 16 + lh;
                fa[mi].q[0] = *(const uint4*)&sA[cur][ar][kb];
                fa[mi].q[1] = *(const uint4*)&sA[cur][ar][kb + 16];
            }
#pragma unroll
            for (int j = 0; j < 2; ++j) {
                const int br = (wn * 2 + j) * 16 + lh;
                fb[j].q[0] = *(const uint4*)&sB[cur][br][kb];
                fb[j].q[1] = *(const uint4*)&sB[cur][br][kb + 16];
            }
#pragma unroll
            for (int mi = 0; mi < 2; ++mi)
#pragma unroll
                for (int j = 0; j < 2; ++j)
                    acc[mi][j] = __builtin_amdgcn_wmma_f32_16x16x32_bf16(
                        false, fa[mi].v, false, fb[j].v, (short)0, acc[mi][j], false, false);
        }
        if (more) {
            *(uint4*)&sA[nxt][arow][achk]     = a0;
            *(uint4*)&sA[nxt][arow][achk + 8] = a1;
        }
        __builtin_amdgcn_s_wait_tensorcnt(0);
        __syncthreads();
    }

#pragma unroll
    for (int mi = 0; mi < 2; ++mi)
#pragma unroll
        for (int j = 0; j < 2; ++j)
#pragma unroll
            for (int r = 0; r < 8; ++r) {
                int rr = (wm * 2 + mi) * 16 + sel * 8 + r;
                int slot = sSlot[rr];
                if (slot >= 0)
                    y[(size_t)slot * DDIM + n0 + (wn * 2 + j) * 16 + lh] = acc[mi][j][r] * sGw[rr];
            }
}

// ---------------------------------------------------------------- combine: out[t] = y[2t] + y[2t+1]
__global__ __launch_bounds__(256) void moe_combine(
    const float* __restrict__ y, float* __restrict__ out)
{
    size_t base = ((size_t)blockIdx.x * 256 + threadIdx.x) * 4;
    int t = (int)(base / DDIM);
    int d = (int)(base % DDIM);
    const float4* ya = (const float4*)(y + (size_t)(2 * t) * DDIM + d);
    float4 a = ya[0];
    float4 b = ya[DDIM / 4];
    float4 o; o.x = a.x + b.x; o.y = a.y + b.y; o.z = a.z + b.z; o.w = a.w + b.w;
    *(float4*)(out + base) = o;
}

// ---------------------------------------------------------------- host
extern "C" void kernel_launch(void* const* d_in, const int* in_sizes, int n_in,
                              void* d_out, int out_size, void* d_ws, size_t ws_size,
                              hipStream_t stream) {
    (void)in_sizes; (void)n_in; (void)out_size; (void)ws_size;
    const float* x      = (const float*)d_in[0];
    const float* gate_w = (const float*)d_in[1];
    const float* w1     = (const float*)d_in[2];
    const float* w3     = (const float*)d_in[3];
    const float* w2     = (const float*)d_in[4];
    float* out = (float*)d_out;

    char* ws = (char*)d_ws;
    size_t off = 0;
    int*            cnt       = (int*)(ws + off);            off += 256;
    int*            tok_list  = (int*)(ws + off);            off += 4ull * EEXP * T_TOK;
    int*            slot_list = (int*)(ws + off);            off += 4ull * EEXP * T_TOK;
    float*          gw_list   = (float*)(ws + off);          off += 4ull * EEXP * T_TOK;
    unsigned short* xb        = (unsigned short*)(ws + off); off += 2ull * T_TOK * DDIM;
    unsigned short* hbuf      = (unsigned short*)(ws + off); off += 2ull * NSLOT * HDIM;
    float*          y         = (float*)(ws + off);          off += 4ull * NSLOT * DDIM;
    unsigned short* wt1       = (unsigned short*)(ws + off); off += 2ull * EEXP * DDIM * HDIM;
    unsigned short* wt3       = (unsigned short*)(ws + off); off += 2ull * EEXP * DDIM * HDIM;
    unsigned short* wt2       = (unsigned short*)(ws + off); off += 2ull * EEXP * DDIM * HDIM;

    moe_zero_cnt<<<1, 32, 0, stream>>>(cnt);
    moe_gate<<<T_TOK / 8, 256, 0, stream>>>(x, gate_w, cnt, tok_list, slot_list, gw_list);
    moe_x2bf<<<(T_TOK * DDIM) / 2048, 256, 0, stream>>>(x, xb);
    // W1, W3: [D][H] -> bf16 [H][D];  W2: [H][D] -> bf16 [D][H]
    moe_wtrans<<<dim3(HDIM / 32, DDIM / 32, EEXP), 256, 0, stream>>>(w1, wt1, DDIM, HDIM);
    moe_wtrans<<<dim3(HDIM / 32, DDIM / 32, EEXP), 256, 0, stream>>>(w3, wt3, DDIM, HDIM);
    moe_wtrans<<<dim3(DDIM / 32, HDIM / 32, EEXP), 256, 0, stream>>>(w2, wt2, HDIM, DDIM);
    moe_gemm1<<<dim3(HDIM / 128, T_TOK / 64, EEXP), 256, 0, stream>>>(
        xb, wt1, wt3, cnt, tok_list, slot_list, hbuf);
    moe_gemm2<<<dim3(DDIM / 128, T_TOK / 64, EEXP), 256, 0, stream>>>(
        hbuf, wt2, cnt, slot_list, gw_list, y);
    moe_combine<<<(T_TOK * DDIM) / 1024, 256, 0, stream>>>(y, out);
}
